// WindowAttention_29085518529112
// MI455X (gfx1250) — matrix operand order repaired and checked by reading, not compiled, not run on hardware
//
#include <hip/hip_runtime.h>

// CDNA5 / gfx1250 WMMA types (wave32)
typedef __attribute__((ext_vector_type(16))) __bf16 v16bf;
typedef __attribute__((ext_vector_type(8)))  __bf16 v8bf;
typedef __attribute__((ext_vector_type(4)))  __bf16 v4bf;
typedef __attribute__((ext_vector_type(4)))  float  v4f;
typedef __attribute__((ext_vector_type(8)))  float  v8f;

#define N_TOK   49
#define HEADS   8
#define HD      32
#define C_DIM   256
#define C3_DIM  768
#define NWIN    64
#define B_WIN   4096
#define M_ROWS  (B_WIN * N_TOK)          // 200704 = 128 * 1568
#define SCALE_Q 0.17677669529663687f     // 32^-0.5

__device__ __forceinline__ v8f wmma_bf16(v16bf a, v16bf b, v8f c) {
    // D = A(16x32) * B(32x16) + C, f32 accumulate
    return __builtin_amdgcn_wmma_f32_16x16x32_bf16(false, a, false, b, (short)0, c, false, false);
}

__device__ __forceinline__ v16bf cat8(v8bf a, v8bf b) {
    return __builtin_shufflevector(a, b, 0, 1, 2, 3, 4, 5, 6, 7,
                                         8, 9, 10, 11, 12, 13, 14, 15);
}

// ---------------------------------------------------------------------------
// Kernel 0: bulk fp32 -> bf16 conversion (RNE fptrunc; native packed cvt).
// ---------------------------------------------------------------------------
__global__ __launch_bounds__(256) void cvt_bf16_kernel(
    const float* __restrict__ in, __bf16* __restrict__ out, int n4)
{
    const int i = blockIdx.x * 256 + threadIdx.x;
    if (i < n4) {
        v4f v = ((const v4f*)in)[i];
        ((v4bf*)out)[i] = __builtin_convertvector(v, v4bf);
    }
}

// ---------------------------------------------------------------------------
// Kernel 1: fused QKV GEMM  (xbf[M,256] @ wbf[768,256]^T + qkv_b)
// Block: 256 thr = 8 waves; block tile M=128 x N=256; wave tile 64x64
// (16 accumulators; 16 loads : 16 WMMA per k-step).
// blockIdx.x = N block (fast-varying) so A strips are reused out of L2.
// ---------------------------------------------------------------------------
__global__ __launch_bounds__(256) void qkv_gemm_kernel(
    const __bf16* __restrict__ x, const __bf16* __restrict__ w,
    const float* __restrict__ bias,
    __bf16* __restrict__ qb, __bf16* __restrict__ kb, __bf16* __restrict__ vb)
{
    const int lane = threadIdx.x & 31;
    const int wid  = threadIdx.x >> 5;
    const int hi   = lane >> 4;          // half-wave select
    const int lm   = lane & 15;
    const int m0 = blockIdx.y * 128 + (wid & 1) * 64;
    const int n0 = blockIdx.x * 256 + (wid >> 1) * 64;

    v8f acc[4][4];
    #pragma unroll
    for (int i = 0; i < 4; i++)
        #pragma unroll
        for (int j = 0; j < 4; j++)
            #pragma unroll
            for (int e = 0; e < 8; e++) acc[i][j][e] = 0.0f;

    for (int k0 = 0; k0 < C_DIM; k0 += 32) {
        v16bf afr[4];
        #pragma unroll
        for (int mt = 0; mt < 4; ++mt) {
            // A 16x32 fragment: lane row = lm, K = e + 16*hi (16 contiguous)
            const __bf16* ap = x + (size_t)(m0 + mt * 16 + lm) * C_DIM + k0 + hi * 16;
            afr[mt] = cat8(*(const v8bf*)ap, *(const v8bf*)(ap + 8));
        }
        v16bf bfr[4];
        #pragma unroll
        for (int nt = 0; nt < 4; ++nt) {
            // B 32x16 fragment: lane col = lm, K = e + 16*hi (16 contiguous, 32B aligned)
            bfr[nt] = *(const v16bf*)(w + (size_t)(n0 + nt * 16 + lm) * C_DIM + k0 + hi * 16);
        }
        #pragma unroll
        for (int mt = 0; mt < 4; mt++)
            #pragma unroll
            for (int nt = 0; nt < 4; nt++)
                acc[mt][nt] = wmma_bf16(afr[mt], bfr[nt], acc[mt][nt]);
    }

    // Per-nt column decomposition (uniform per 16-col tile)
    int   which[4], hoff[4];
    float bv[4];
    #pragma unroll
    for (int nt = 0; nt < 4; nt++) {
        const int col = n0 + nt * 16 + lm;
        which[nt] = col >> 8;
        hoff[nt]  = ((col >> 5) & 7) * (N_TOK * HD) + (col & 31);  // h*1568 + d
        bv[nt]    = bias[col];
    }
    // Epilogue: C layout row = base + r + 8*hi, col = lm.
    #pragma unroll
    for (int mt = 0; mt < 4; mt++) {
        #pragma unroll
        for (int r = 0; r < 8; r++) {
            const int row = m0 + mt * 16 + r + hi * 8;
            const int bw  = row / N_TOK;
            const int tok = row - bw * N_TOK;
            const size_t rb = (size_t)bw * (HEADS * N_TOK * HD) + tok * HD;
            #pragma unroll
            for (int nt = 0; nt < 4; nt++) {
                const size_t off = rb + hoff[nt];
                const float val = acc[mt][nt][r] + bv[nt];
                if (which[nt] == 0)      qb[off] = (__bf16)(val * SCALE_Q);
                else if (which[nt] == 1) kb[off] = (__bf16)val;
                else                     vb[off] = (__bf16)val;
            }
        }
    }
}

// ---------------------------------------------------------------------------
// Kernel 2: fused attention per (window, head).
// Block = 128 thr = 4 waves; tokens padded 49->64 (4 M-tiles).
// ---------------------------------------------------------------------------
__global__ __launch_bounds__(128) void attn_kernel(
    const __bf16* __restrict__ qb, const __bf16* __restrict__ kb,
    const __bf16* __restrict__ vb,
    const float* __restrict__ mask, const float* __restrict__ bias_table,
    const int* __restrict__ rel_idx,
    __bf16* __restrict__ ob)
{
    __shared__ __bf16 qs[64][32];
    __shared__ __bf16 ks[64][32];
    __shared__ __bf16 vts[32][64];   // v transposed: [d][token]
    __shared__ __bf16 ps[64][64];    // softmaxed probabilities

    const int b = blockIdx.x, h = blockIdx.y;
    const int tid  = threadIdx.x;
    const int lane = tid & 31, wid = tid >> 5;
    const int hi = lane >> 4, lm = lane & 15;

    const size_t base = (size_t)(b * HEADS + h) * N_TOK * HD;
    const __bf16* qg = qb + base;
    const __bf16* kg = kb + base;
    const __bf16* vg = vb + base;

    v8bf z8;
    #pragma unroll
    for (int i = 0; i < 8; i++) z8[i] = (__bf16)0.0f;

    // Stage Q,K (zero-padded rows 49..63) and V transposed into LDS.
    for (int chunk = tid; chunk < 256; chunk += 128) {
        const int row = chunk >> 2, c8 = (chunk & 3) * 8;
        v8bf qv = z8, kv = z8, vv = z8;
        if (row < N_TOK) {
            qv = *(const v8bf*)(qg + row * HD + c8);
            kv = *(const v8bf*)(kg + row * HD + c8);
            vv = *(const v8bf*)(vg + row * HD + c8);
        }
        *(v8bf*)&qs[row][c8] = qv;
        *(v8bf*)&ks[row][c8] = kv;
        #pragma unroll
        for (int j = 0; j < 8; j++) vts[c8 + j][row] = vv[j];
    }
    __syncthreads();

    // S = Q * K^T for this wave's 16-row strip. K dim = 32 -> single WMMA/tile.
    v16bf qf;
    {
        const __bf16* qr = &qs[wid * 16 + lm][0];
        qf = cat8(*(const v8bf*)(qr + hi * 16), *(const v8bf*)(qr + hi * 16 + 8));
    }
    v8f s[4];
    #pragma unroll
    for (int nt = 0; nt < 4; nt++) {
        const v16bf kf = *(const v16bf*)&ks[nt * 16 + lm][hi * 16];  // 32B aligned
        v8f zc;
        #pragma unroll
        for (int e = 0; e < 8; e++) zc[e] = 0.0f;
        s[nt] = wmma_bf16(qf, kf, zc);
    }

    // bias gather + shift-mask add + numerically stable softmax per row.
    const float* mp = mask + (size_t)(b & (NWIN - 1)) * N_TOK * N_TOK;
    float sv[4][8];
    float rsum[8];
    #pragma unroll
    for (int r = 0; r < 8; r++) {
        const int row_t = wid * 16 + r + hi * 8;
        float m = -1e30f;
        #pragma unroll
        for (int nt = 0; nt < 4; nt++) {
            const int col_t = nt * 16 + lm;
            float v = -1e30f;
            if (row_t < N_TOK && col_t < N_TOK) {
                const int ri = rel_idx[row_t * N_TOK + col_t];
                v = s[nt][r] + bias_table[ri * HEADS + h] + mp[row_t * N_TOK + col_t];
            }
            sv[nt][r] = v;
            m = fmaxf(m, v);
        }
        #pragma unroll
        for (int xm = 8; xm >= 1; xm >>= 1) m = fmaxf(m, __shfl_xor(m, xm, 32));
        float sum = 0.0f;
        #pragma unroll
        for (int nt = 0; nt < 4; nt++) {
            const float e = __expf(sv[nt][r] - m);
            sv[nt][r] = e;
            sum += e;
        }
        #pragma unroll
        for (int xm = 8; xm >= 1; xm >>= 1) sum += __shfl_xor(sum, xm, 32);
        rsum[r] = sum;
    }
    #pragma unroll
    for (int r = 0; r < 8; r++) {
        const float inv = 1.0f / rsum[r];
        const int row = wid * 16 + r + hi * 8;
        #pragma unroll
        for (int nt = 0; nt < 4; nt++)
            ps[row][nt * 16 + lm] = (__bf16)(sv[nt][r] * inv);
    }
    __syncthreads();

    // O = P * V : M strip 16 x K 64 (2 chunks) x N 32 (2 tiles).
    v8f o[2];
    #pragma unroll
    for (int i = 0; i < 2; i++)
        #pragma unroll
        for (int e = 0; e < 8; e++) o[i][e] = 0.0f;

    #pragma unroll
    for (int kc = 0; kc < 2; kc++) {
        const __bf16* pr = &ps[wid * 16 + lm][kc * 32];
        const v16bf pf = cat8(*(const v8bf*)(pr + hi * 16), *(const v8bf*)(pr + hi * 16 + 8));
        #pragma unroll
        for (int nt = 0; nt < 2; nt++) {
            const v16bf vf = *(const v16bf*)&vts[nt * 16 + lm][kc * 32 + hi * 16];
            o[nt] = wmma_bf16(pf, vf, o[nt]);
        }
    }

    // attn-out layout for proj GEMM: [b*49+tok, h*32+d] bf16
    #pragma unroll
    for (int nt = 0; nt < 2; nt++) {
        const int d = nt * 16 + lm;
        #pragma unroll
        for (int r = 0; r < 8; r++) {
            const int tok = wid * 16 + r + hi * 8;
            if (tok < N_TOK)
                ob[(size_t)(b * N_TOK + tok) * C_DIM + h * HD + d] = (__bf16)o[nt][r];
        }
    }
}

// ---------------------------------------------------------------------------
// Kernel 3: proj GEMM  (attn_out(bf16)[M,256] @ pwbf[256,256]^T + proj_b) -> f32
// Block tile M=128 x N=256; wave tile 64x64.
// ---------------------------------------------------------------------------
__global__ __launch_bounds__(256) void proj_gemm_kernel(
    const __bf16* __restrict__ a, const __bf16* __restrict__ w,
    const float* __restrict__ bias, float* __restrict__ out)
{
    const int lane = threadIdx.x & 31;
    const int wid  = threadIdx.x >> 5;
    const int hi = lane >> 4, lm = lane & 15;
    const int m0 = blockIdx.x * 128 + (wid & 1) * 64;
    const int n0 = (wid >> 1) * 64;

    v8f acc[4][4];
    #pragma unroll
    for (int i = 0; i < 4; i++)
        #pragma unroll
        for (int j = 0; j < 4; j++)
            #pragma unroll
            for (int e = 0; e < 8; e++) acc[i][j][e] = 0.0f;

    for (int k0 = 0; k0 < C_DIM; k0 += 32) {
        v16bf afr[4];
        #pragma unroll
        for (int mt = 0; mt < 4; ++mt) {
            const __bf16* ap = a + (size_t)(m0 + mt * 16 + lm) * C_DIM + k0 + hi * 16;
            afr[mt] = cat8(*(const v8bf*)ap, *(const v8bf*)(ap + 8));
        }
        v16bf bfr[4];
        #pragma unroll
        for (int nt = 0; nt < 4; ++nt) {
            bfr[nt] = *(const v16bf*)(w + (size_t)(n0 + nt * 16 + lm) * C_DIM + k0 + hi * 16);
        }
        #pragma unroll
        for (int mt = 0; mt < 4; mt++)
            #pragma unroll
            for (int nt = 0; nt < 4; nt++)
                acc[mt][nt] = wmma_bf16(afr[mt], bfr[nt], acc[mt][nt]);
    }

    #pragma unroll
    for (int nt = 0; nt < 4; nt++) {
        const int col  = n0 + nt * 16 + lm;
        const float bv = bias[col];
        #pragma unroll
        for (int mt = 0; mt < 4; mt++) {
            #pragma unroll
            for (int r = 0; r < 8; r++) {
                const int row = m0 + mt * 16 + r + hi * 8;
                out[(size_t)row * C_DIM + col] = acc[mt][nt][r] + bv;
            }
        }
    }
}

// ---------------------------------------------------------------------------
extern "C" void kernel_launch(void* const* d_in, const int* in_sizes, int n_in,
                              void* d_out, int out_size, void* d_ws, size_t ws_size,
                              hipStream_t stream) {
    const float* x      = (const float*)d_in[0];
    const float* mask   = (const float*)d_in[1];
    const float* qkv_w  = (const float*)d_in[2];
    const float* qkv_b  = (const float*)d_in[3];
    const float* proj_w = (const float*)d_in[4];
    const float* proj_b = (const float*)d_in[5];
    const float* btab   = (const float*)d_in[6];
    const int*   ridx   = (const int*)d_in[7];
    float* out = (float*)d_out;

    // Workspace layout (bf16):
    //   Q, K, V, attn-out : each B*H*49*32 = 51,380,224 elems
    //   xbf               : M*256        = 51,380,224 elems
    //   wqbf              : 768*256      = 196,608 elems
    //   wpbf              : 256*256      = 65,536 elems
    const size_t per = (size_t)B_WIN * HEADS * N_TOK * HD;
    __bf16* qb   = (__bf16*)d_ws;
    __bf16* kb   = qb + per;
    __bf16* vb   = kb + per;
    __bf16* ab   = vb + per;
    __bf16* xbf  = ab + per;
    __bf16* wqbf = xbf + (size_t)M_ROWS * C_DIM;
    __bf16* wpbf = wqbf + (size_t)C3_DIM * C_DIM;

    // Pre-pass conversions (all sizes divisible by 1024)
    cvt_bf16_kernel<<<dim3((M_ROWS * C_DIM) / 1024), 256, 0, stream>>>(x, xbf, (M_ROWS * C_DIM) / 4);
    cvt_bf16_kernel<<<dim3((C3_DIM * C_DIM) / 1024), 256, 0, stream>>>(qkv_w, wqbf, (C3_DIM * C_DIM) / 4);
    cvt_bf16_kernel<<<dim3((C_DIM * C_DIM) / 1024), 256, 0, stream>>>(proj_w, wpbf, (C_DIM * C_DIM) / 4);

    // N on blockIdx.x (fast) -> the 3 N-blocks sharing an A strip run adjacently (L2 reuse)
    qkv_gemm_kernel<<<dim3(C3_DIM / 256, M_ROWS / 128), 256, 0, stream>>>(
        xbf, wqbf, qkv_b, qb, kb, vb);
    attn_kernel<<<dim3(B_WIN, HEADS), 128, 0, stream>>>(
        qb, kb, vb, mask, btab, ridx, ab);
    proj_gemm_kernel<<<dim3(M_ROWS / 128), 256, 0, stream>>>(
        ab, wpbf, proj_b, out);
}
